// Head_Relative_Positions_21758304321794
// MI455X (gfx1250) — matrix operand order repaired and checked by reading, not compiled, not run on hardware
//
#include <hip/hip_runtime.h>
#include <hip/hip_bf16.h>
#include <stdint.h>

// ---------------------------------------------------------------------------
// CDNA5 WMMA (wave32): D(16x16 f32) = A(16x32 bf16) x B(32x16 bf16) + C
// Fragments built as pure 32-bit vector ops (two 16B loads -> shuffle ->
// bitcast) so no 16-bit element scalarization is generated.
// ---------------------------------------------------------------------------
typedef __bf16    v16bf  __attribute__((ext_vector_type(16)));
typedef float     v8f    __attribute__((ext_vector_type(8)));
typedef unsigned  uint4v __attribute__((ext_vector_type(4)));
typedef unsigned  uint8v __attribute__((ext_vector_type(8)));

__device__ __forceinline__ v16bf mkfrag(uint4v lo, uint4v hi) {
  uint8v w = __builtin_shufflevector(lo, hi, 0, 1, 2, 3, 4, 5, 6, 7);
  return __builtin_bit_cast(v16bf, w);
}
__device__ __forceinline__ v8f wmma_bf16(v16bf a, v16bf b, v8f c) {
  return __builtin_amdgcn_wmma_f32_16x16x32_bf16(false, a, false, b, (short)0, c, false, false);
}

__device__ __forceinline__ unsigned short f2bf(float f) {
  unsigned u = __float_as_uint(f);
  unsigned r = u + 0x7FFFu + ((u >> 16) & 1u);   // round-to-nearest-even
  return (unsigned short)(r >> 16);
}
__device__ __forceinline__ float bf2f(unsigned short h) {
  return __uint_as_float(((unsigned)h) << 16);
}

__device__ __forceinline__ void lds_fence() {
  asm volatile("s_wait_dscnt 0" ::: "memory");
}
#define WAIT_ASYNC(n) asm volatile("s_wait_asynccnt " #n ::: "memory")

// Async global->LDS copy, 16B per lane (GVS mode: SGPR64 base + VGPR32 offset).
__device__ __forceinline__ void async_g2l_b128(unsigned ldsaddr, const void* base, int voff) {
  asm volatile("global_load_async_to_lds_b128 %0, %1, %2"
               :
               : "v"(ldsaddr), "v"(voff), "s"((unsigned long long)(uintptr_t)base)
               : "memory");
}

// ---------------------------------------------------------------------------
// Problem constants
// ---------------------------------------------------------------------------
#define BB 256
#define TT 256
#define CC 384
#define HH 64
#define NROW (BB * TT)
#define WPAIR_PER_MAT (CC / 2 * HH)    // 12288 dwords
#define WS_QOFF (3 * WPAIR_PER_MAT * 4)
#define BFBUF ((size_t)NROW * HH)

// ---------------------------------------------------------------------------
// Kernel 0: pack weights fp32 -> bf16 pairs, fragment-contiguous layout:
//   dword index = (((mat*12 + kk)*2 + hf)*64 + n)*8 + j
//   holds { bf16(W[2kp][n]), bf16(W[2kp+1][n]) },  kp = 16*kk + 8*hf + j
// ---------------------------------------------------------------------------
__global__ void prep_weights(const float* __restrict__ Wq, const float* __restrict__ Wk,
                             const float* __restrict__ Wv, unsigned* __restrict__ wpair) {
  int tid = blockIdx.x * blockDim.x + threadIdx.x;   // 0 .. 36863
  int j  = tid & 7;
  int n  = (tid >> 3) & 63;
  int hf = (tid >> 9) & 1;
  int r  = tid >> 10;
  int kk = r % 12;
  int mat = r / 12;
  int kp = 16 * kk + 8 * hf + j;
  const float* W = (mat == 0) ? Wq : (mat == 1) ? Wk : Wv;
  unsigned lo = f2bf(W[(2 * kp) * HH + n]);
  unsigned hi = f2bf(W[(2 * kp + 1) * HH + n]);
  wpair[tid] = lo | (hi << 16);
}

// ---------------------------------------------------------------------------
// Kernel 1: q/k/v projections. 4 waves/block, 16 rows of x per wave.
// q,k row-major bf16; v TRANSPOSED per batch: vT[b][h][t] (16B packed stores).
// ---------------------------------------------------------------------------
__global__ void __launch_bounds__(128)
qkv_kernel(const float* __restrict__ x, const unsigned* __restrict__ wpair,
           unsigned short* __restrict__ qbf, unsigned short* __restrict__ kbf,
           unsigned short* __restrict__ vbf) {
  __shared__ __align__(16) unsigned short xs[4][16][392];   // 50176 B, 784B rows

  const int lane = threadIdx.x & 31;
  const int wv   = threadIdx.x >> 5;
  const int rowbase = (blockIdx.x * 4 + wv) * 16;
  const int mrow = lane & 15;
  const int hf   = lane >> 4;

  // Stage 16x384 fp32 -> bf16 LDS with float4 loads + b64 LDS stores.
  const float4* x4 = (const float4*)x;
  for (int i = 0; i < 48; ++i) {
    int idx = i * 32 + lane;
    int row = idx / 96;
    int c4  = idx - row * 96;
    float4 f = x4[(size_t)(rowbase + row) * 96 + c4];
    uint2 pk;
    pk.x = (unsigned)f2bf(f.x) | ((unsigned)f2bf(f.y) << 16);
    pk.y = (unsigned)f2bf(f.z) | ((unsigned)f2bf(f.w) << 16);
    *(uint2*)&xs[wv][row][c4 * 4] = pk;
  }
  lds_fence();

  // 12 A-fragments in registers; each = two b128 LDS reads.
  v16bf afr[12];
#pragma unroll
  for (int kk = 0; kk < 12; ++kk) {
    uint4v lo = *(const uint4v*)&xs[wv][mrow][32 * kk + 16 * hf];
    uint4v hi = *(const uint4v*)&xs[wv][mrow][32 * kk + 16 * hf + 8];
    afr[kk] = mkfrag(lo, hi);
  }

  const uint4v* w4 = (const uint4v*)wpair;
#pragma unroll
  for (int mat = 0; mat < 3; ++mat) {
#pragma unroll
    for (int nt = 0; nt < 4; ++nt) {
      // per-(mat,nt) weight fragment base; kk advances by 256 uint4s (imm offs)
      const uint4v* wp = w4 + (size_t)(((mat * 24 + hf) * 64) + 16 * nt + mrow) * 2;
      v8f acc = {0.f, 0.f, 0.f, 0.f, 0.f, 0.f, 0.f, 0.f};
      for (int kk = 0; kk < 12; ++kk) {
        v16bf bfr = mkfrag(wp[(size_t)kk * 256], wp[(size_t)kk * 256 + 1]);
        acc = wmma_bf16(afr[kk], bfr, acc);
      }
      if (mat < 2) {
        unsigned short* dst = ((mat == 0) ? qbf : kbf) +
                              (size_t)(rowbase + 8 * hf) * HH + 16 * nt + mrow;
#pragma unroll
        for (int r = 0; r < 8; ++r)
          dst[r * HH] = f2bf(acc[r]);        // imm offsets r*128B
      } else {
        uint4 pk;   // vT: lane holds 8 consecutive t -> one 16B store
        pk.x = (unsigned)f2bf(acc[0]) | ((unsigned)f2bf(acc[1]) << 16);
        pk.y = (unsigned)f2bf(acc[2]) | ((unsigned)f2bf(acc[3]) << 16);
        pk.z = (unsigned)f2bf(acc[4]) | ((unsigned)f2bf(acc[5]) << 16);
        pk.w = (unsigned)f2bf(acc[6]) | ((unsigned)f2bf(acc[7]) << 16);
        int bidx = rowbase >> 8;
        int tb   = (rowbase & 255) + 8 * hf;
        *(uint4*)&vbf[((size_t)bidx * HH + 16 * nt + mrow) * TT + tb] = pk;
      }
    }
  }
}

// ---------------------------------------------------------------------------
// Kernel 2: causal attention + relative positions. ONE wave per block.
// Score rows in LDS (single-pass softmax, 32-lane split + shfl_xor combine).
// exp() results written as bf16 into P -> P.V A-frags are two b128 reads.
// Q/K tiles and vT chunks staged with double-buffered async-to-LDS.
// ---------------------------------------------------------------------------
__global__ void __launch_bounds__(32)
attn_kernel(const unsigned short* __restrict__ qbf, const unsigned short* __restrict__ kbf,
            const unsigned short* __restrict__ vbf, const float* __restrict__ relk,
            const float* __restrict__ relv, float* __restrict__ out) {
  __shared__ float S[16][257];                              // 16448 B (fp32 scores)
  __shared__ __align__(16) unsigned short P[16][264];       //  8448 B (bf16 probs)
  __shared__ __align__(16) unsigned short stg[2][16][72];   //  4608 B (144B rows)
  __shared__ __align__(16) unsigned short vstg[2][64][40];  // 10240 B (80B rows)
  __shared__ float qrL[16][4];
  __shared__ float coefL[16][4];
  __shared__ float invL[16];

  const int lane = threadIdx.x;
  const int b    = blockIdx.x >> 4;
  const int t0   = (blockIdx.x & 15) << 4;
  const int mrow = lane & 15;
  const int hf   = lane >> 4;
  const unsigned stg0  = (unsigned)(uintptr_t)&stg[0][0][0];
  const unsigned stg1  = (unsigned)(uintptr_t)&stg[1][0][0];
  const unsigned vstg0 = (unsigned)(uintptr_t)&vstg[0][0][0];
  const unsigned vstg1 = (unsigned)(uintptr_t)&vstg[1][0][0];

  // ---- Stage Q tile (2048B) into stg buf0 ----
  const unsigned short* qsrc = qbf + (size_t)(b * TT + t0) * HH;
  for (int it = 0; it < 4; ++it) {
    int o = it * 512 + lane * 16;
    async_g2l_b128(stg0 + (unsigned)((o >> 7) * 144 + (o & 127)), qsrc, o);
  }
  WAIT_ASYNC(0);
  lds_fence();

  // Q A-fragments (head dim K = 2 x 32); two b128 reads each.
  v16bf aq[2];
#pragma unroll
  for (int f = 0; f < 2; ++f) {
    uint4v lo = *(const uint4v*)&stg[0][mrow][32 * f + 8 * hf];
    uint4v hi = *(const uint4v*)&stg[0][mrow][32 * f + 16 + 8 * hf];
    aq[f] = mkfrag(lo, hi);
  }

  // qr[row][j] = q[row] . rel_k_table[j] (j=0..3), h-range split across halves
  {
    float a0 = 0.f, a1 = 0.f, a2 = 0.f, a3 = 0.f;
    for (int h = hf * 32; h < hf * 32 + 32; ++h) {
      float qv = bf2f(stg[0][mrow][h]);
      a0 += qv * relk[h];
      a1 += qv * relk[HH + h];
      a2 += qv * relk[2 * HH + h];
      a3 += qv * relk[3 * HH + h];
    }
    a0 += __shfl_xor(a0, 16);
    a1 += __shfl_xor(a1, 16);
    a2 += __shfl_xor(a2, 16);
    a3 += __shfl_xor(a3, 16);
    if (lane < 16) {
      qrL[mrow][0] = a0; qrL[mrow][1] = a1; qrL[mrow][2] = a2; qrL[mrow][3] = a3;
    }
  }
  lds_fence();

  const int ntile = (t0 >> 4) + 1;
  // Prologue: stage K tile 0 into buf1.
  {
    const unsigned short* ksrc = kbf + (size_t)(b * TT) * HH;
    for (int it = 0; it < 4; ++it) {
      int o = it * 512 + lane * 16;
      async_g2l_b128(stg1 + (unsigned)((o >> 7) * 144 + (o & 127)), ksrc, o);
    }
  }

  // ---- S = Q K^T, double-buffered K tiles ----
  for (int i = 0; i < ntile; ++i) {
    const int cb = (i + 1) & 1;
    if (i + 1 < ntile) {
      lds_fence();
      const unsigned short* ksrc = kbf + (size_t)(b * TT + (i + 1) * 16) * HH;
      unsigned dstb = (i & 1) ? stg1 : stg0;
      for (int it = 0; it < 4; ++it) {
        int o = it * 512 + lane * 16;
        async_g2l_b128(dstb + (unsigned)((o >> 7) * 144 + (o & 127)), ksrc, o);
      }
      WAIT_ASYNC(4);
    } else {
      WAIT_ASYNC(0);
    }

    v8f sacc = {0.f, 0.f, 0.f, 0.f, 0.f, 0.f, 0.f, 0.f};
#pragma unroll
    for (int f = 0; f < 2; ++f) {
      uint4v lo = *(const uint4v*)&stg[cb][mrow][32 * f + 8 * hf];
      uint4v hi = *(const uint4v*)&stg[cb][mrow][32 * f + 16 + 8 * hf];
      sacc = wmma_bf16(aq[f], mkfrag(lo, hi), sacc);
    }
#pragma unroll
    for (int r = 0; r < 8; ++r)
      S[r + 8 * hf][i * 16 + mrow] = sacc[r];
  }
  lds_fence();

  const int rounds32 = (ntile + 1) >> 1;
  const int scols    = rounds32 * 32;
  const int t        = t0 + mrow;

  // ---- Softmax: 32-lane split (lane pair (l, l+16) shares a row) ----
  {
    float q0 = qrL[mrow][0], q1 = qrL[mrow][1], q2 = qrL[mrow][2], q3 = qrL[mrow][3];
    int sbeg = hf * 128;
    int send = (t < sbeg + 127) ? t : (sbeg + 127);        // inclusive
    float mx = -3.4e38f;
    for (int s = sbeg; s <= send; ++s) {
      int d = t - s;
      float rel = (d >= 3) ? q0 : (d == 2) ? q1 : (d == 1) ? q2 : q3;
      float vv = (S[mrow][s] + rel) * 0.125f;              // 1/sqrt(64)
      S[mrow][s] = vv;
      mx = fmaxf(mx, vv);
    }
    mx = fmaxf(mx, __shfl_xor(mx, 16));
    lds_fence();
    float sum = 0.f;
    for (int s = sbeg; s <= send; ++s) {
      float p = __expf(S[mrow][s] - mx);
      P[mrow][s] = f2bf(p);
      sum += p;
    }
    sum += __shfl_xor(sum, 16);
    int zb = (sbeg > t + 1) ? sbeg : (t + 1);
    int ze = (sbeg + 128 < scols) ? (sbeg + 128) : scols;
    for (int s = zb; s < ze; ++s) P[mrow][s] = 0;
    lds_fence();
    if (lane < 16) {
      float d3 = __expf(S[mrow][t] - mx);
      float d2 = (t >= 1) ? __expf(S[mrow][t - 1] - mx) : 0.f;
      float d1 = (t >= 2) ? __expf(S[mrow][t - 2] - mx) : 0.f;
      float inv = 1.f / sum;
      coefL[mrow][0] = (sum - d1 - d2 - d3) * inv;
      coefL[mrow][1] = d1 * inv;
      coefL[mrow][2] = d2 * inv;
      coefL[mrow][3] = d3 * inv;
      invL[mrow] = inv;
    }
  }
  lds_fence();

  // ---- O = P V (unnormalized), vT chunks of 32 s-values, double-buffered ----
  v8f accO[4];
#pragma unroll
  for (int nt = 0; nt < 4; ++nt) accO[nt] = (v8f){0.f, 0.f, 0.f, 0.f, 0.f, 0.f, 0.f, 0.f};

  const char* vbase = (const char*)vbf;
  {
    const char* vsrc = vbase + (size_t)b * HH * TT * 2;
    for (int it = 0; it < 8; ++it) {
      int o = it * 512 + lane * 16;
      int h = o >> 6, c = o & 63;
      async_g2l_b128(vstg0 + (unsigned)(h * 80 + c), vsrc, h * 512 + c);
    }
  }
  for (int f2 = 0; f2 < rounds32; ++f2) {
    const int vb = f2 & 1;
    if (f2 + 1 < rounds32) {
      lds_fence();
      const char* vsrc = vbase + ((size_t)b * HH * TT + (f2 + 1) * 32) * 2;
      unsigned dstb = vb ? vstg0 : vstg1;
      for (int it = 0; it < 8; ++it) {
        int o = it * 512 + lane * 16;
        int h = o >> 6, c = o & 63;
        async_g2l_b128(dstb + (unsigned)(h * 80 + c), vsrc, h * 512 + c);
      }
      WAIT_ASYNC(8);
    } else {
      WAIT_ASYNC(0);
    }

    int s0 = f2 * 32;
    v16bf ap = mkfrag(*(const uint4v*)&P[mrow][s0 + 8 * hf],
                      *(const uint4v*)&P[mrow][s0 + 16 + 8 * hf]);
#pragma unroll
    for (int nt = 0; nt < 4; ++nt) {
      int h = 16 * nt + mrow;
      v16bf bv = mkfrag(*(const uint4v*)&vstg[vb][h][8 * hf],
                        *(const uint4v*)&vstg[vb][h][16 + 8 * hf]);
      accO[nt] = wmma_bf16(ap, bv, accO[nt]);
    }
  }

  // ---- Epilogue: out = inv*O + sum_j coef_j * rel_v[j] ----
  float invr[8], c0r[8], c1r[8], c2r[8], c3r[8];
#pragma unroll
  for (int r = 0; r < 8; ++r) {
    int row = r + 8 * hf;
    invr[r] = invL[row];
    c0r[r]  = coefL[row][0];
    c1r[r]  = coefL[row][1];
    c2r[r]  = coefL[row][2];
    c3r[r]  = coefL[row][3];
  }
  float* obase = out + ((size_t)b * TT + t0 + 8 * hf) * HH + mrow;
#pragma unroll
  for (int nt = 0; nt < 4; ++nt) {
    int h = 16 * nt + mrow;
    float rv0 = relv[h], rv1 = relv[HH + h], rv2 = relv[2 * HH + h], rv3 = relv[3 * HH + h];
#pragma unroll
    for (int r = 0; r < 8; ++r) {
      obase[r * HH + 16 * nt] =          // imm offsets, same lane layout as accO
          invr[r] * accO[nt][r] + c0r[r] * rv0 + c1r[r] * rv1 + c2r[r] * rv2 + c3r[r] * rv3;
    }
  }
}

// ---------------------------------------------------------------------------
// Host launcher. Workspace (25.3 MB):
//   [0, 147456)   packed bf16 weight pairs (fragment-contiguous)
//   [+8MB] x3     q (row-major), k (row-major), v (transposed per batch)
// ---------------------------------------------------------------------------
extern "C" void kernel_launch(void* const* d_in, const int* in_sizes, int n_in,
                              void* d_out, int out_size, void* d_ws, size_t ws_size,
                              hipStream_t stream) {
  (void)in_sizes; (void)n_in; (void)out_size; (void)ws_size;
  const float* x    = (const float*)d_in[0];
  const float* Wq   = (const float*)d_in[1];
  const float* Wk   = (const float*)d_in[2];
  const float* Wv   = (const float*)d_in[3];
  const float* relk = (const float*)d_in[4];
  const float* relv = (const float*)d_in[5];
  float* out = (float*)d_out;

  char* ws = (char*)d_ws;
  unsigned*       wpair = (unsigned*)ws;
  unsigned short* qbf   = (unsigned short*)(ws + WS_QOFF);
  unsigned short* kbf   = qbf + BFBUF;
  unsigned short* vbf   = kbf + BFBUF;

  prep_weights<<<dim3(144), dim3(256), 0, stream>>>(Wq, Wk, Wv, wpair);
  qkv_kernel<<<dim3(1024), dim3(128), 0, stream>>>(x, wpair, qbf, kbf, vbf);
  attn_kernel<<<dim3(4096), dim3(32), 0, stream>>>(qbf, kbf, vbf, relk, relv, out);
}
